// LSTMCRF_10574209483155
// MI455X (gfx1250) — matrix operand order repaired and checked by reading, not compiled, not run on hardware
//
#include <hip/hip_runtime.h>
#include <hip/hip_bf16.h>
#include <math.h>

#define B_   128
#define T_   512
#define E_   768
#define H_   256
#define L_   9
#define BT_  (B_ * T_)
#define G4H  (4 * H_)   // 1024

typedef __attribute__((ext_vector_type(16))) __bf16 v16bf;
typedef __attribute__((ext_vector_type(8)))  float  v8f;

// ---------- helpers ----------

__device__ __forceinline__ unsigned short f2bf(float f) {
    union { float f; unsigned int u; } x; x.f = f;
    unsigned int r = x.u + 0x7FFFu + ((x.u >> 16) & 1u);   // round-nearest-even
    return (unsigned short)(r >> 16);
}
__device__ __forceinline__ float bf2f(unsigned short h) {
    union { float f; unsigned int u; } x; x.u = ((unsigned int)h) << 16;
    return x.f;
}
__device__ __forceinline__ float sigm(float x) { return 1.0f / (1.0f + __expf(-x)); }

__device__ __forceinline__ v8f wmma_bf16(v16bf a, v16bf b, v8f c) {
    return __builtin_amdgcn_wmma_f32_16x16x32_bf16(false, a, false, b, (short)0, c,
                                                   false, false);
}

// A fragment, 16x32 bf16, M rows from `base` (row-major, leading dim lda).
// lanes 0-15: elems 0-7 = K[kt..kt+7], elems 8-15 = K[kt+16..kt+23]
// lanes 16-31: +8 on both K ranges.
__device__ __forceinline__ v16bf load_a_frag(const unsigned short* base, int lda,
                                             int mbase, int kt) {
    int lane  = threadIdx.x & 31;
    int mrow  = lane & 15;
    int khalf = lane >> 4;
    const unsigned short* p = base + (size_t)(mbase + mrow) * lda + kt + 8 * khalf;
    union { v16bf v; uint4 q[2]; } f;
    f.q[0] = *(const uint4*)(p);
    f.q[1] = *(const uint4*)(p + 16);
    return f.v;
}

// B fragment, 32x16 bf16, B[k][n] = W[n][k] with W row-major [N,K] (leading dim ldw).
// lanes 0-15: elems 0-7 = K[kt..kt+7], elems 8-15 = K[kt+16..kt+23]; lanes 16-31: +8; col = lane&15.
// Works for global or LDS (generic) pointers.
__device__ __forceinline__ v16bf load_b_frag(const unsigned short* w, int ldw,
                                             int nbase, int kt) {
    int lane  = threadIdx.x & 31;
    int ncol  = lane & 15;
    int khalf = lane >> 4;
    const unsigned short* p = w + (size_t)(nbase + ncol) * ldw + kt + 8 * khalf;
    union { v16bf v; uint4 q[2]; } f;
    f.q[0] = *(const uint4*)(p);
    f.q[1] = *(const uint4*)(p + 16);
    return f.v;
}

// ---------- kernels ----------

__global__ void f2bf_kernel(const float* __restrict__ src,
                            unsigned short* __restrict__ dst, size_t n) {
    size_t i = (size_t)blockIdx.x * blockDim.x + threadIdx.x;
    size_t stride = (size_t)gridDim.x * blockDim.x;
    for (; i < n; i += stride) dst[i] = f2bf(src[i]);
}

// xW = x (bf16 [BT,E]) * Wih^T (bf16 [4H,E]) + (bih+bhh), fp32 out [BT,4H].
// Block: 8 waves = 8 M-tiles (128 rows) sharing ONE 128-col N-slice.
// B k-chunks (8KB) double-buffered in LDS via async global->LDS DMA; all waves
// read identical LDS fragments (broadcast), cutting L2 B-traffic 8x.
__global__ __launch_bounds__(256) void gemm_xw(
    const unsigned short* __restrict__ xbf,
    const unsigned short* __restrict__ wih_f, const unsigned short* __restrict__ wih_b,
    const float* __restrict__ bih_f, const float* __restrict__ bhh_f,
    const float* __restrict__ bih_b, const float* __restrict__ bhh_b,
    float* __restrict__ xw_f, float* __restrict__ xw_b) {
    const int dir = blockIdx.z;
    const unsigned short* W = dir ? wih_b : wih_f;
    const float* b1 = dir ? bih_b : bih_f;
    const float* b2 = dir ? bhh_b : bhh_f;
    float* out = dir ? xw_b : xw_f;

    const int wv    = threadIdx.x >> 5;
    const int mbase = blockIdx.x * 128 + wv * 16;  // per-wave M tile
    const int nbase = blockIdx.y * 128;            // shared N slice (8 tiles)

    __shared__ __align__(16) unsigned short sB[2][128 * 32];  // [buf][n_local][k_local]

    // stage one 128x32 bf16 chunk of W (rows nbase..+127, k kt..kt+31) into sB[buf]
    auto stage = [&](int buf, int kt) {
        int tid = threadIdx.x;
#pragma unroll
        for (int j = 0; j < 2; j++) {
            int idx = tid + j * 256;   // 0..511 b128 chunks
            int r = idx >> 2;          // local row
            int c = idx & 3;           // 16B chunk within row
            unsigned long long g =
                (unsigned long long)(size_t)(const void*)(W + (size_t)(nbase + r) * E_ + kt + c * 8);
            unsigned lo = (unsigned)(size_t)(const void*)(&sB[buf][r * 32 + c * 8]);
            asm volatile("global_load_async_to_lds_b128 %0, %1, off"
                         :: "v"(lo), "v"(g) : "memory");
        }
    };

    v8f zero = {0.f, 0.f, 0.f, 0.f, 0.f, 0.f, 0.f, 0.f};
    v8f acc[8];
#pragma unroll
    for (int t = 0; t < 8; t++) acc[t] = zero;

    const int nk = E_ / 32;  // 24
    stage(0, 0);
    for (int i = 0; i < nk; i++) {
        int kt = i * 32;
        if (i + 1 < nk) {
            stage((i + 1) & 1, kt + 32);
            asm volatile("s_wait_asynccnt 0x2" ::: "memory");  // chunk i landed
        } else {
            asm volatile("s_wait_asynccnt 0x0" ::: "memory");
        }
        __syncthreads();  // all waves' staging of chunk i visible
        v16bf a = load_a_frag(xbf, E_, mbase, kt);
        const unsigned short* sb = &sB[i & 1][0];
#pragma unroll
        for (int t = 0; t < 8; t++) {
            v16bf b = load_b_frag(sb, 32, t * 16, 0);
            acc[t] = wmma_bf16(a, b, acc[t]);
        }
        __syncthreads();  // done reading buf before it is restaged
    }

    const int lane = threadIdx.x & 31;
    const int col  = lane & 15;
    const int half = lane >> 4;
#pragma unroll
    for (int t = 0; t < 8; t++) {
        int n = nbase + t * 16 + col;
        float bias = b1[n] + b2[n];
#pragma unroll
        for (int r = 0; r < 8; r++) {
            int m = mbase + r + 8 * half;
            out[(size_t)m * G4H + n] = acc[t][r] + bias;
        }
    }
}

// Persistent recurrent LSTM. blockIdx.x = batch group of 16 rows, blockIdx.y = dir.
// Wave wv owns j in [wv*32, wv*32+32) for ALL 4 gates -> LSTM elementwise update
// operates directly on its own WMMA D fragments; h (bf16, 8KB) lives in LDS.
// K in [0,128) of Whh (256KB for all 1024 gate rows) is staged ONCE in dynamic LDS
// (CDNA5 320KB/WG); K in [128,256) streams from L2 -> serial L2 traffic halved.
__global__ __launch_bounds__(256) void lstm_rec(
    const float* __restrict__ xw_f, const float* __restrict__ xw_b,
    const unsigned short* __restrict__ whh_f, const unsigned short* __restrict__ whh_b,
    unsigned short* __restrict__ hg_f, unsigned short* __restrict__ hg_b) {
    const int dir = blockIdx.y;
    const float* xw = dir ? xw_b : xw_f;
    const unsigned short* W = dir ? whh_b : whh_f;
    unsigned short* hout = dir ? hg_b : hg_f;
    const int bbase = blockIdx.x * 16;

    extern __shared__ unsigned short dlds[];
    unsigned short* whh_sh = dlds;                  // [1024][128] bf16 = 256KB
    unsigned short* h_sh   = dlds + 1024 * 128;     // [16][256]  bf16 = 8KB

    const int wv   = threadIdx.x >> 5;
    const int lane = threadIdx.x & 31;
    const int col  = lane & 15;
    const int half = lane >> 4;
    const int j0   = wv * 32;

    // zero h, stage Whh[:, 0:128) in per-wave tile-major row order:
    // local row = wv*128 + tt*16 + ncol with tt = jt*4+g  ->  n = g*256 + wv*32 + jt*16 + ncol
    for (int i = threadIdx.x; i < 16 * 256; i += 256) h_sh[i] = 0;
    for (int idx = threadIdx.x; idx < 1024 * 16; idx += 256) {  // b128 chunks
        int lr = idx >> 4;
        int kc = idx & 15;
        int lwv = lr >> 7, rem = lr & 127;
        int tt = rem >> 4, ncol = rem & 15;
        int g = tt & 3, jt = tt >> 2;
        int n = g * 256 + lwv * 32 + jt * 16 + ncol;
        *(uint4*)(whh_sh + (size_t)lr * 128 + kc * 8) =
            *(const uint4*)(W + (size_t)n * H_ + kc * 8);
    }

    float c_reg[2][8];
#pragma unroll
    for (int jt = 0; jt < 2; jt++)
#pragma unroll
        for (int r = 0; r < 8; r++) c_reg[jt][r] = 0.f;

    __syncthreads();

    for (int step = 0; step < T_; step++) {
        const int t = dir ? (T_ - 1 - step) : step;

        v8f acc[2][4];  // [j-tile][gate i,f,g,o]
#pragma unroll
        for (int jt = 0; jt < 2; jt++)
#pragma unroll
            for (int g = 0; g < 4; g++) {
                int n = g * 256 + j0 + jt * 16 + col;
#pragma unroll
                for (int r = 0; r < 8; r++) {
                    int m = r + 8 * half;
                    acc[jt][g][r] = xw[((size_t)(bbase + m) * T_ + t) * G4H + n];
                }
            }

        // K in [0,128): B fragments from LDS-staged Whh
        for (int kt = 0; kt < 128; kt += 32) {
            v16bf a = load_a_frag(h_sh, 256, 0, kt);
#pragma unroll
            for (int jt = 0; jt < 2; jt++)
#pragma unroll
                for (int g = 0; g < 4; g++) {
                    v16bf b = load_b_frag(whh_sh, 128,
                                          wv * 128 + (jt * 4 + g) * 16, kt);
                    acc[jt][g] = wmma_bf16(a, b, acc[jt][g]);
                }
        }
        // K in [128,256): B fragments stream from global (L2)
        for (int kt = 128; kt < H_; kt += 32) {
            v16bf a = load_a_frag(h_sh, 256, 0, kt);
#pragma unroll
            for (int jt = 0; jt < 2; jt++)
#pragma unroll
                for (int g = 0; g < 4; g++) {
                    v16bf b = load_b_frag(W, H_, g * 256 + j0 + jt * 16, kt);
                    acc[jt][g] = wmma_bf16(a, b, acc[jt][g]);
                }
        }

        __syncthreads();  // everyone done reading h(t-1)

#pragma unroll
        for (int jt = 0; jt < 2; jt++)
#pragma unroll
            for (int r = 0; r < 8; r++) {
                float zi = acc[jt][0][r], zf = acc[jt][1][r];
                float zg = acc[jt][2][r], zo = acc[jt][3][r];
                float c = sigm(zf) * c_reg[jt][r] + sigm(zi) * tanhf(zg);
                float h = sigm(zo) * tanhf(c);
                c_reg[jt][r] = c;
                int m = r + 8 * half;
                int j = j0 + jt * 16 + col;
                unsigned short hb = f2bf(h);
                h_sh[m * 256 + j] = hb;
                hout[((size_t)(bbase + m) * T_ + t) * H_ + j] = hb;
            }

        __syncthreads();  // h(t) visible before next step's reads
    }
}

// emissions[bt, l] = [h_f|h_b] . W_out[l,:] + b_out[l]   (N=9 -> scalar kernel)
__global__ void emis_kernel(const unsigned short* __restrict__ h_f,
                            const unsigned short* __restrict__ h_b,
                            const float* __restrict__ W_out,
                            const float* __restrict__ b_out,
                            float* __restrict__ em) {
    int idx = blockIdx.x * blockDim.x + threadIdx.x;
    if (idx >= BT_ * L_) return;
    int l  = idx % L_;
    int bt = idx / L_;
    const unsigned short* hf = h_f + (size_t)bt * H_;
    const unsigned short* hb = h_b + (size_t)bt * H_;
    const float* w = W_out + (size_t)l * (2 * H_);
    float acc = b_out[l];
    for (int j = 0; j < H_; j++) acc += bf2f(hf[j]) * w[j];
    for (int j = 0; j < H_; j++) acc += bf2f(hb[j]) * w[H_ + j];
    em[idx] = acc;
}

// CRF gold-path score + forward algorithm; one thread per batch element.
__global__ void crf_kernel(const float* __restrict__ em,
                           const int* __restrict__ mask,
                           const int* __restrict__ labels,
                           const float* __restrict__ start_t,
                           const float* __restrict__ end_t,
                           const float* __restrict__ trans,
                           float* __restrict__ partial) {
    __shared__ float tr[L_ * L_];
    if (threadIdx.x < L_ * L_) tr[threadIdx.x] = trans[threadIdx.x];
    __syncthreads();
    int b = threadIdx.x;
    if (b >= B_) return;

    const float* emb = em + (size_t)b * T_ * L_;
    const int* mb = mask + (size_t)b * T_;
    const int* lb = labels + (size_t)b * T_;

    // numerator
    int prev = lb[0];
    float score = start_t[prev] + emb[prev];
    for (int t = 1; t < T_; t++) {
        if (mb[t]) {
            int lt = lb[t];
            score += emb[t * L_ + lt] + tr[prev * L_ + lt];
            prev = lt;
        }
    }
    score += end_t[prev];

    // denominator: forward algorithm
    float alpha[L_];
#pragma unroll
    for (int j = 0; j < L_; j++) alpha[j] = start_t[j] + emb[j];
    for (int t = 1; t < T_; t++) {
        if (!mb[t]) continue;
        float nxt[L_];
#pragma unroll
        for (int j = 0; j < L_; j++) {
            float mx = -3.4e38f;
#pragma unroll
            for (int i = 0; i < L_; i++)
                mx = fmaxf(mx, alpha[i] + tr[i * L_ + j]);
            float s = 0.f;
#pragma unroll
            for (int i = 0; i < L_; i++)
                s += __expf(alpha[i] + tr[i * L_ + j] - mx);
            nxt[j] = mx + __logf(s) + emb[t * L_ + j];
        }
#pragma unroll
        for (int j = 0; j < L_; j++) alpha[j] = nxt[j];
    }
    float mx = -3.4e38f;
#pragma unroll
    for (int j = 0; j < L_; j++) mx = fmaxf(mx, alpha[j] + end_t[j]);
    float s = 0.f;
#pragma unroll
    for (int j = 0; j < L_; j++) s += __expf(alpha[j] + end_t[j] - mx);
    float logZ = mx + __logf(s);

    partial[b] = score - logZ;
}

__global__ void reduce_kernel(const float* __restrict__ partial, float* __restrict__ out) {
    if (threadIdx.x == 0) {
        float s = 0.f;
        for (int b = 0; b < B_; b++) s += partial[b];
        out[0] = -s;
    }
}

// ---------- launcher ----------

extern "C" void kernel_launch(void* const* d_in, const int* in_sizes, int n_in,
                              void* d_out, int out_size, void* d_ws, size_t ws_size,
                              hipStream_t stream) {
    const float* x       = (const float*)d_in[0];
    const int*   mask    = (const int*)d_in[1];
    const int*   labels  = (const int*)d_in[2];
    const float* Wih_f   = (const float*)d_in[3];
    const float* Whh_f   = (const float*)d_in[4];
    const float* bih_f   = (const float*)d_in[5];
    const float* bhh_f   = (const float*)d_in[6];
    const float* Wih_b   = (const float*)d_in[7];
    const float* Whh_b   = (const float*)d_in[8];
    const float* bih_b   = (const float*)d_in[9];
    const float* bhh_b   = (const float*)d_in[10];
    const float* W_out   = (const float*)d_in[11];
    const float* b_out   = (const float*)d_in[12];
    const float* start_t = (const float*)d_in[13];
    const float* end_t   = (const float*)d_in[14];
    const float* trans   = (const float*)d_in[15];
    float* out = (float*)d_out;

    char* ws = (char*)d_ws;
    size_t off = 0;
    auto carve = [&](size_t bytes) -> void* {
        void* p = ws + off;
        off += (bytes + 255) & ~(size_t)255;
        return p;
    };
    unsigned short* x_bf    = (unsigned short*)carve((size_t)BT_ * E_ * 2);
    unsigned short* wihf_bf = (unsigned short*)carve((size_t)G4H * E_ * 2);
    unsigned short* wihb_bf = (unsigned short*)carve((size_t)G4H * E_ * 2);
    unsigned short* whhf_bf = (unsigned short*)carve((size_t)G4H * H_ * 2);
    unsigned short* whhb_bf = (unsigned short*)carve((size_t)G4H * H_ * 2);
    float*          xw_f    = (float*)carve((size_t)BT_ * G4H * 4);
    float*          xw_b    = (float*)carve((size_t)BT_ * G4H * 4);
    unsigned short* h_f     = (unsigned short*)carve((size_t)BT_ * H_ * 2);
    unsigned short* h_b     = (unsigned short*)carve((size_t)BT_ * H_ * 2);
    float*          emis    = (float*)carve((size_t)BT_ * L_ * 4);
    float*          part    = (float*)carve((size_t)B_ * 4);
    (void)ws_size; (void)n_in; (void)in_sizes; (void)out_size;

    // 1) fp32 -> bf16 conversions
    f2bf_kernel<<<2048, 256, 0, stream>>>(x, x_bf, (size_t)BT_ * E_);
    f2bf_kernel<<<512, 256, 0, stream>>>(Wih_f, wihf_bf, (size_t)G4H * E_);
    f2bf_kernel<<<512, 256, 0, stream>>>(Wih_b, wihb_bf, (size_t)G4H * E_);
    f2bf_kernel<<<256, 256, 0, stream>>>(Whh_f, whhf_bf, (size_t)G4H * H_);
    f2bf_kernel<<<256, 256, 0, stream>>>(Whh_b, whhb_bf, (size_t)G4H * H_);

    // 2) big WMMA GEMM with async-LDS staged B, both directions
    gemm_xw<<<dim3(BT_ / 128, G4H / 128, 2), 256, 0, stream>>>(
        x_bf, wihf_bf, wihb_bf, bih_f, bhh_f, bih_b, bhh_b, xw_f, xw_b);

    // 3) persistent recurrent LSTM (WMMA per step; 256KB of Whh LDS-resident)
    size_t rec_lds = (size_t)(1024 * 128 + 16 * 256) * sizeof(unsigned short);
    lstm_rec<<<dim3(B_ / 16, 2), 256, rec_lds, stream>>>(
        xw_f, xw_b, whhf_bf, whhb_bf, h_f, h_b);

    // 4) emissions projection (L=9 -> scalar)
    emis_kernel<<<(BT_ * L_ + 255) / 256, 256, 0, stream>>>(
        h_f, h_b, W_out, b_out, emis);

    // 5) CRF per-batch numerator + forward algorithm
    crf_kernel<<<1, 128, 0, stream>>>(emis, mask, labels, start_t, end_t, trans, part);

    // 6) final negative-sum
    reduce_kernel<<<1, 32, 0, stream>>>(part, out);
}
